// Net_82394652606805
// MI455X (gfx1250) — compile-verified, matrix operand order checked by reading, not run on hardware
//
#include <hip/hip_runtime.h>
#include <hip/hip_bf16.h>

// ---------------------------------------------------------------------------
// LSTM (S=512,B=256,IN=128,H=256) + FC head for MI455X (gfx1250, wave32, WMMA)
//
//  * bf16 WMMA (v_wmma_f32_16x16x32_bf16) with fp32 accumulate.
//  * A = [h_{t-1} | x_t] (K=384), B = [W_hh | W_ih]^T slice per block.
//  * Persistent kernel, 8 blocks (32 hidden cols each), global atomic barrier
//    per step; cell state c lives in VGPRs for all 512 steps (gates of one
//    hidden unit land in the same lane of the C fragments).
//  * B fragments staged in 96KB LDS once (CDNA5 WGP has 320KB) -> ds_load_b128
//    per step instead of the register-spill storm of the fully-unrolled form.
//  * h ping-pongs between two global buffers in WMMA-A fragment layout.
// ---------------------------------------------------------------------------

typedef __bf16 v16bf __attribute__((ext_vector_type(16)));
typedef float  v8f   __attribute__((ext_vector_type(8)));

#define S_LEN  512
#define BATCH  256
#define IN_DIM 128
#define HID    256
#define NBLK   8      // persistent blocks; each owns 32 hidden columns
#define CLS    10

#define WPB_BYTES (8 * 12 * 32 * 32)   // 98304 B of B-fragments per block

// ---- workspace layout (bytes) ---------------------------------------------
static constexpr size_t OFF_WPACK = 0;
static constexpr size_t SZ_WPACK  = (size_t)NBLK * WPB_BYTES;            // 768 KB
static constexpr size_t OFF_XPACK = OFF_WPACK + SZ_WPACK;
static constexpr size_t SZ_XPACK  = (size_t)S_LEN * 16 * 4 * 32 * 32;    // 32 MB
static constexpr size_t OFF_HP0   = OFF_XPACK + SZ_XPACK;
static constexpr size_t SZ_HP     = (size_t)16 * 8 * 32 * 32;            // 128 KB
static constexpr size_t OFF_HP1   = OFF_HP0 + SZ_HP;
static constexpr size_t OFF_HLAST = OFF_HP1 + SZ_HP;
static constexpr size_t SZ_HLAST  = (size_t)BATCH * HID * 4;
static constexpr size_t OFF_BAR   = OFF_HLAST + SZ_HLAST;
// total ~34.9 MB

// (lane, elem) -> K index inside a 32-wide K tile, per ISA 16-bit A/B layout
// (CDNA5 ISA 7.12.2).
__device__ __host__ __forceinline__ int kmap(int lane, int e) {
  int v = e >> 1, w = e & 1;
  int base = (lane < 16) ? 0 : 8;
  return (v < 4) ? (base + 2 * v + w) : (16 + base + 2 * (v - 4) + w);
}

__device__ __forceinline__ float fsig(float x) {
  return 1.0f / (1.0f + __expf(-x));
}

__device__ __forceinline__ v8f v8f_zero() {
  v8f z;
#pragma unroll
  for (int i = 0; i < 8; ++i) z[i] = 0.0f;
  return z;
}

// ---------------------------------------------------------------------------
__global__ void k_init(uint4* __restrict__ hp0, unsigned* __restrict__ bar) {
  int t = blockIdx.x * blockDim.x + threadIdx.x;
  if (t < (int)(SZ_HP / 16)) hp0[t] = make_uint4(0u, 0u, 0u, 0u);
  if (t == 0) *bar = 0u;
}

// Pack [W_hh | W_ih] into per-block WMMA-B fragments (bf16).
__global__ void k_pack_w(const float* __restrict__ W_ih,
                         const float* __restrict__ W_hh,
                         __hip_bfloat16* __restrict__ wp) {
  int t = blockIdx.x * blockDim.x + threadIdx.x;   // 393216 total
  int e    = t & 15;
  int l    = (t >> 4) & 31;
  int kt   = (t >> 9) % 12;
  int nt   = (t / 6144) & 7;
  int blk  = t / 49152;
  int n_local = nt * 16 + (l & 15);                // 0..127 within block
  int gate = n_local >> 5;                         // i,f,g,o
  int jj   = n_local & 31;
  int r    = gate * HID + blk * 32 + jj;
  int k    = kt * 32 + kmap(l, e);                 // 0..383
  float v  = (k < HID) ? W_hh[r * HID + k] : W_ih[r * IN_DIM + (k - HID)];
  wp[t] = __float2bfloat16(v);
}

// Pack x into per-step WMMA-A fragments (bf16).
__global__ void k_pack_x(const float* __restrict__ x,
                         __hip_bfloat16* __restrict__ xp) {
  size_t t = (size_t)blockIdx.x * blockDim.x + threadIdx.x;  // 16777216 total
  int e   = (int)(t & 15);
  int l   = (int)((t >> 4) & 31);
  int kt2 = (int)((t >> 9) & 3);
  int mt  = (int)((t >> 11) & 15);
  int s   = (int)(t >> 15);
  int m   = mt * 16 + (l & 15);                    // batch index
  int k   = kt2 * 32 + kmap(l, e);                 // input-dim index 0..127
  xp[t] = __float2bfloat16(x[((size_t)s * BATCH + m) * IN_DIM + k]);
}

// ---------------------------------------------------------------------------
// Persistent recurrent kernel: 8 blocks x 256 threads (8 waves).
// Wave w owns batch rows [32w, 32w+32): 2 M-tiles x 8 N-tiles of C.
__global__ void __launch_bounds__(256, 1)
k_lstm(const __hip_bfloat16* __restrict__ wp,
       const __hip_bfloat16* __restrict__ xp,
       __hip_bfloat16* __restrict__ hp0,
       __hip_bfloat16* __restrict__ hp1,
       const float* __restrict__ b_ih,
       const float* __restrict__ b_hh,
       float* __restrict__ hlast,
       unsigned* __restrict__ bar) {
  extern __shared__ char smem[];      // 96KB: this block's B fragments

  const int blk  = blockIdx.x;        // 0..7 -> hidden cols [32*blk, 32*blk+32)
  const int tid  = threadIdx.x;
  const int wv   = tid >> 5;          // wave id 0..7
  const int lane = tid & 31;
  const int n_in = lane & 15;

  // ---- stage this block's weight fragments into LDS (once) ---------------
  {
    const uint4* src = (const uint4*)(wp + (size_t)blk * (WPB_BYTES / 2));
    uint4* dst = (uint4*)smem;
#pragma unroll 4
    for (int i = tid; i < WPB_BYTES / 16; i += 256) dst[i] = src[i];
  }

  // per-thread gate biases: bias[gate][njtile]
  float bias[4][2];
#pragma unroll
  for (int g = 0; g < 4; ++g)
#pragma unroll
    for (int nj = 0; nj < 2; ++nj) {
      int r = g * HID + blk * 32 + nj * 16 + n_in;
      bias[g][nj] = b_ih[r] + b_hh[r];
    }

  __syncthreads();  // LDS weights visible to all waves

  // cell state, resident in VGPRs for the whole sequence
  v8f c[2][2];
#pragma unroll
  for (int i = 0; i < 2; ++i)
#pragma unroll
    for (int j = 0; j < 2; ++j) c[i][j] = v8f_zero();

  for (int s = 0; s < S_LEN; ++s) {
    const __hip_bfloat16* hin  = (s & 1) ? hp1 : hp0;
    __hip_bfloat16*       hout = (s & 1) ? hp0 : hp1;
    const __hip_bfloat16* xs   = xp + (size_t)s * 32768;

    if (s + 1 < S_LEN)  // warm L2 for next step's x fragments
      __builtin_prefetch(xp + (size_t)(s + 1) * 32768 + wv * 4096 + lane * 16, 0, 1);

    v8f acc[2][8];
#pragma unroll
    for (int i = 0; i < 2; ++i)
#pragma unroll
      for (int j = 0; j < 8; ++j) acc[i][j] = v8f_zero();

    // ---- K loop, h part (K = 0..255): A from global hin, B from LDS -------
#pragma unroll 1
    for (int kt = 0; kt < 8; ++kt) {
      v16bf a0 = *(const v16bf*)(hin + ((wv * 2 + 0) * 8 + kt) * 512 + lane * 16);
      v16bf a1 = *(const v16bf*)(hin + ((wv * 2 + 1) * 8 + kt) * 512 + lane * 16);
#pragma unroll
      for (int nt = 0; nt < 8; ++nt) {
        v16bf b = *(const v16bf*)(smem + (nt * 12 + kt) * 1024 + lane * 32);
        acc[0][nt] = __builtin_amdgcn_wmma_f32_16x16x32_bf16(
            false, a0, false, b, (short)0, acc[0][nt], false, false);
        acc[1][nt] = __builtin_amdgcn_wmma_f32_16x16x32_bf16(
            false, a1, false, b, (short)0, acc[1][nt], false, false);
      }
    }
    // ---- K loop, x part (K = 256..383): A from global xs, B from LDS ------
#pragma unroll 1
    for (int kt2 = 0; kt2 < 4; ++kt2) {
      v16bf a0 = *(const v16bf*)(xs + ((wv * 2 + 0) * 4 + kt2) * 512 + lane * 16);
      v16bf a1 = *(const v16bf*)(xs + ((wv * 2 + 1) * 4 + kt2) * 512 + lane * 16);
#pragma unroll
      for (int nt = 0; nt < 8; ++nt) {
        v16bf b = *(const v16bf*)(smem + (nt * 12 + (8 + kt2)) * 1024 + lane * 32);
        acc[0][nt] = __builtin_amdgcn_wmma_f32_16x16x32_bf16(
            false, a0, false, b, (short)0, acc[0][nt], false, false);
        acc[1][nt] = __builtin_amdgcn_wmma_f32_16x16x32_bf16(
            false, a1, false, b, (short)0, acc[1][nt], false, false);
      }
    }

    // ---- LSTM cell epilogue (lane-local: i/f/g/o share lane & vgpr) -------
#pragma unroll
    for (int mti = 0; mti < 2; ++mti) {
#pragma unroll
      for (int nj = 0; nj < 2; ++nj) {
#pragma unroll
        for (int v = 0; v < 8; ++v) {
          float iv = fsig(acc[mti][nj    ][v] + bias[0][nj]);
          float fv = fsig(acc[mti][nj + 2][v] + bias[1][nj]);
          float gv = tanhf(acc[mti][nj + 4][v] + bias[2][nj]);
          float ov = fsig(acc[mti][nj + 6][v] + bias[3][nj]);
          float cv = fv * c[mti][nj][v] + iv * gv;
          c[mti][nj][v] = cv;
          float hv = ov * tanhf(cv);

          // scatter h (bf16) into next step's WMMA-A fragment layout
          int m_in = v + ((lane < 16) ? 0 : 8);
          int m    = wv * 32 + mti * 16 + m_in;     // batch row
          int jj   = nj * 16 + n_in;                // 0..31, K tile = blk
          int la   = ((jj & 8) ? 16 : 0) + (m & 15);
          int kkp  = jj & ~8;
          int vv   = (kkp < 8) ? (kkp >> 1) : (4 + ((kkp - 16) >> 1));
          int e    = vv * 2 + (kkp & 1);
          hout[((m >> 4) * 8 + blk) * 512 + la * 16 + e] = __float2bfloat16(hv);
          if (s == S_LEN - 1) hlast[m * HID + blk * 32 + jj] = hv;
        }
      }
    }

    // ---- device-wide step barrier (monotonic counter) ---------------------
    __threadfence();
    __syncthreads();
    if (tid == 0) {
      atomicAdd(bar, 1u);
      unsigned target = (unsigned)(s + 1) * NBLK;
      while (*(volatile unsigned*)bar < target) __builtin_amdgcn_s_sleep(1);
    }
    __syncthreads();
    __threadfence();
  }
}

// ---------------------------------------------------------------------------
__global__ void k_fc(const float* __restrict__ hlast,
                     const float* __restrict__ W_fc,
                     float* __restrict__ out) {
  int cls = blockIdx.x;    // 0..9
  int b   = threadIdx.x;   // 0..255
  float s = 0.0f;
#pragma unroll 4
  for (int j = 0; j < HID; ++j)
    s = fmaf(hlast[b * HID + j], W_fc[cls * HID + j], s);
  out[b * CLS + cls] = s;
}

// ---------------------------------------------------------------------------
extern "C" void kernel_launch(void* const* d_in, const int* in_sizes, int n_in,
                              void* d_out, int out_size, void* d_ws,
                              size_t ws_size, hipStream_t stream) {
  const float* x    = (const float*)d_in[0];
  const float* W_ih = (const float*)d_in[1];
  const float* W_hh = (const float*)d_in[2];
  const float* b_ih = (const float*)d_in[3];
  const float* b_hh = (const float*)d_in[4];
  const float* W_fc = (const float*)d_in[5];

  char* ws = (char*)d_ws;
  __hip_bfloat16* wp    = (__hip_bfloat16*)(ws + OFF_WPACK);
  __hip_bfloat16* xpk   = (__hip_bfloat16*)(ws + OFF_XPACK);
  __hip_bfloat16* hp0   = (__hip_bfloat16*)(ws + OFF_HP0);
  __hip_bfloat16* hp1   = (__hip_bfloat16*)(ws + OFF_HP1);
  float*          hlast = (float*)(ws + OFF_HLAST);
  unsigned*       bar   = (unsigned*)(ws + OFF_BAR);

  // zero h0 + barrier counter (ws is poisoned by the harness)
  k_init<<<32, 256, 0, stream>>>((uint4*)hp0, bar);
  // pack weights and x into bf16 WMMA fragment layouts
  k_pack_w<<<1536, 256, 0, stream>>>(W_ih, W_hh, wp);
  k_pack_x<<<65536, 256, 0, stream>>>(x, xpk);
  // persistent recurrence: 8 blocks, 96KB dynamic LDS for weight fragments
  k_lstm<<<NBLK, 256, WPB_BYTES, stream>>>(wp, xpk, hp0, hp1, b_ih, b_hh,
                                           hlast, bar);
  // final classifier head
  k_fc<<<CLS, 256, 0, stream>>>(hlast, W_fc, (float*)d_out);
}
